// GNNViolenceDetector_78658031059125
// MI455X (gfx1250) — compile-verified, matrix operand order checked by reading, not run on hardware
//
#include <hip/hip_runtime.h>
#include <hip/hip_bf16.h>
#include <math.h>

// ---------------------------------------------------------------------------
// Problem constants (from the reference)
// ---------------------------------------------------------------------------
#define TT    1024          // frames
#define NPG   32            // nodes per graph
#define NTOT  (TT * NPG)    // 32768 nodes
#define EE    (NTOT * 8)    // 262144 edges
#define HID   256

typedef __attribute__((ext_vector_type(4)))  __bf16 v4bf;
typedef __attribute__((ext_vector_type(8)))  __bf16 v8bf;
typedef __attribute__((ext_vector_type(16))) __bf16 v16bf;
typedef __attribute__((ext_vector_type(4)))  float  v4f;
typedef __attribute__((ext_vector_type(8)))  float  v8f;

#define SROW 40   // LDS row stride (bf16 elems): 80B, 20 banks -> conflict-free b128

// Fragment permutation: swap bits 3 and 4 of k (involution). Storing tiles with
// columns permuted by kperm makes each lane's 16-element WMMA fragment a single
// contiguous 32B run: positions [0..15] = k{0..7,16..23} (lane half 0),
// positions [16..31] = k{8..15,24..31} (lane half 1).
__device__ __forceinline__ int kperm(int k) {
    return (k & ~24) | ((k & 8) << 1) | ((k & 16) >> 1);
}

__device__ __forceinline__ v16bf frag_load(const __bf16* p) {
    v8bf lo = *(const v8bf*)p;
    v8bf hi = *(const v8bf*)(p + 8);
    return __builtin_shufflevector(lo, hi, 0, 1, 2, 3, 4, 5, 6, 7,
                                   8, 9, 10, 11, 12, 13, 14, 15);
}

// ---------------------------------------------------------------------------
// WMMA GEMM:  C[M,N] = A[M,K] @ op(B) (+ bias[n])
//   op(B) = B[K,N] row-major (b_trans=0) or B[N,K]^T (b_trans=1)
// Block tile 64(M) x 64(N), K stepped by 32. 8 waves (2x4); each wave owns two
// 16x16 WMMA tiles stacked in M (shared B fragment -> 2 v_wmma per K step).
// Operands staged to LDS as bf16 in fragment-permuted order.
// ---------------------------------------------------------------------------
__global__ __launch_bounds__(256) void k_gemm_bf16(
    const float* __restrict__ A, const float* __restrict__ B,
    const float* __restrict__ bias, float* __restrict__ C,
    int M, int N, int K, int lda, int ldb, int ldc, int b_trans)
{
    __shared__ __bf16 sA[64][SROW];   // [m][perm(k)]
    __shared__ __bf16 sB[64][SROW];   // [n][perm(k)]
    const int tid  = threadIdx.x;
    const int m0   = blockIdx.x * 64;
    const int n0   = blockIdx.y * 64;
    const int wave = tid >> 5, lane = tid & 31;
    const int wm   = wave & 1, wn = wave >> 1;   // 2x4 wave grid
    const int l16  = lane & 15, lh = lane >> 4;

    const bool mnfull = (m0 + 64 <= M) && (n0 + 64 <= N);

    v8f acc0 = {};
    v8f acc1 = {};
    for (int k0 = 0; k0 < K; k0 += 32) {
        if (mnfull && (k0 + 32 <= K)) {
            // ---------- fast path: vectorized, branch-free staging ----------
#pragma unroll
            for (int it = 0; it < 2; ++it) {            // A: 64x32 f32
                int i = (tid + it * 256) * 4;
                int r = i >> 5, c = i & 31;
                v4f v = *(const v4f*)&A[(size_t)(m0 + r) * lda + k0 + c];
                v4bf t;
                t[0] = (__bf16)v[0]; t[1] = (__bf16)v[1];
                t[2] = (__bf16)v[2]; t[3] = (__bf16)v[3];
                *(v4bf*)&sA[r][kperm(c)] = t;
            }
            if (b_trans) {
#pragma unroll
                for (int it = 0; it < 2; ++it) {        // B: 64(n)x32(k), k contiguous
                    int i = (tid + it * 256) * 4;
                    int n = i >> 5, c = i & 31;
                    v4f v = *(const v4f*)&B[(size_t)(n0 + n) * ldb + k0 + c];
                    v4bf t;
                    t[0] = (__bf16)v[0]; t[1] = (__bf16)v[1];
                    t[2] = (__bf16)v[2]; t[3] = (__bf16)v[3];
                    *(v4bf*)&sB[n][kperm(c)] = t;
                }
            } else {
#pragma unroll
                for (int it = 0; it < 2; ++it) {        // B: 32(k)x64(n), n contiguous
                    int i = (tid + it * 256) * 4;
                    int r = i >> 6, c = i & 63;
                    v4f v = *(const v4f*)&B[(size_t)(k0 + r) * ldb + n0 + c];
                    int pr = kperm(r);
                    sB[c + 0][pr] = (__bf16)v[0];
                    sB[c + 1][pr] = (__bf16)v[1];
                    sB[c + 2][pr] = (__bf16)v[2];
                    sB[c + 3][pr] = (__bf16)v[3];
                }
            }
            if (k0 + 32 < K) {  // speculative prefetch of next A tile
                int gm = m0 + (tid >> 2);
                int gk = k0 + 32 + ((tid & 3) << 3);
                __builtin_prefetch(&A[(size_t)gm * lda + gk], 0, 1);
            }
        } else {
            // ---------- guarded fallback (K tails of the two tiny GEMMs) ----
            for (int i = tid; i < 64 * 32; i += 256) {
                int r = i >> 5, c = i & 31;
                int gm = m0 + r, gk = k0 + c;
                float v = (gm < M && gk < K) ? A[(size_t)gm * lda + gk] : 0.f;
                sA[r][kperm(c)] = (__bf16)v;
            }
            for (int i = tid; i < 64 * 32; i += 256) {
                int n = i >> 5, c = i & 31;
                int gn = n0 + n, gk = k0 + c;
                float v = 0.f;
                if (gk < K && gn < N)
                    v = b_trans ? B[(size_t)gn * ldb + gk] : B[(size_t)gk * ldb + gn];
                sB[n][kperm(c)] = (__bf16)v;
            }
        }
        __syncthreads();

        v16bf av0 = frag_load(&sA[wm * 32 + l16][lh * 16]);
        v16bf av1 = frag_load(&sA[wm * 32 + 16 + l16][lh * 16]);
        v16bf bv  = frag_load(&sB[wn * 16 + l16][lh * 16]);
        acc0 = __builtin_amdgcn_wmma_f32_16x16x32_bf16(
            false, av0, false, bv, (short)0, acc0, false, false);
        acc1 = __builtin_amdgcn_wmma_f32_16x16x32_bf16(
            false, av1, false, bv, (short)0, acc1, false, false);
        __syncthreads();
    }

    // store: VGPR r holds M = r (lanes 0-15) / r+8 (lanes 16-31)
    int gn = n0 + wn * 16 + l16;
    float badd = (bias && gn < N) ? bias[gn] : 0.f;
#pragma unroll
    for (int r = 0; r < 8; ++r) {
        int gm0 = m0 + wm * 32 + r + 8 * lh;
        int gm1 = gm0 + 16;
        if (gm0 < M && gn < N) C[(size_t)gm0 * ldc + gn] = acc0[r] + badd;
        if (gm1 < M && gn < N) C[(size_t)gm1 * ldc + gn] = acc1[r] + badd;
    }
}

// ---------------------------------------------------------------------------
// Per-node attention coefficients: es[n,h] = sum_c H[n,h,c]*a_s[h,c], same ed.
// ---------------------------------------------------------------------------
__global__ __launch_bounds__(128) void k_attn_coef(
    const float* __restrict__ Hb, const float* __restrict__ a_s,
    const float* __restrict__ a_d, float* __restrict__ es,
    float* __restrict__ ed, int heads)
{
    int n = blockIdx.x;
    int c = threadIdx.x;
    __shared__ float tmp_s[4], tmp_d[4];
    for (int hd = 0; hd < heads; ++hd) {
        float hv = Hb[(size_t)n * heads * 128 + hd * 128 + c];
        float ps = hv * a_s[hd * 128 + c];
        float pd = hv * a_d[hd * 128 + c];
        for (int o = 16; o > 0; o >>= 1) {
            ps += __shfl_down(ps, o, 32);
            pd += __shfl_down(pd, o, 32);
        }
        int w = c >> 5;
        if ((c & 31) == 0) { tmp_s[w] = ps; tmp_d[w] = pd; }
        __syncthreads();
        if (c == 0) {
            es[(size_t)n * heads + hd] = tmp_s[0] + tmp_s[1] + tmp_s[2] + tmp_s[3];
            ed[(size_t)n * heads + hd] = tmp_d[0] + tmp_d[1] + tmp_d[2] + tmp_d[3];
        }
        __syncthreads();
    }
}

// ---------------------------------------------------------------------------
// Per-graph dense GAT softmax + aggregation + bias + BN(+ReLU).
// One block (256 threads) per graph. The graph's whole H block (32 x F, which
// is contiguous in memory) is staged into LDS so the 32-term aggregation dots
// run out of LDS. Dense 32x32 count-weighted attention (exact for dup edges).
// ---------------------------------------------------------------------------
__global__ __launch_bounds__(256) void k_gat_edge(
    const float* __restrict__ Hb, const int* __restrict__ src,
    const int* __restrict__ dst, const float* __restrict__ es,
    const float* __restrict__ ed, const float* __restrict__ bias,
    const float* __restrict__ gamma, const float* __restrict__ beta,
    float* __restrict__ Xout, int heads, int do_relu)
{
    const int g   = blockIdx.x;
    const int tid = threadIdx.x;
    const int F   = heads * 128;

    __shared__ float Hloc[NPG * 512];          // up to 64KB (fits 320KB WGP LDS)
    __shared__ int   cnt[NPG * NPG];
    __shared__ float esL[NPG][4], edL[NPG][4];
    __shared__ float mL[NPG][4], wsL[NPG][4];
    __shared__ float alpha[NPG * NPG][4];

    // stage H block (contiguous NPG*F floats), vectorized
    {
        const float* hp = Hb + (size_t)g * NPG * F;
        int nv = (NPG * F) >> 2;
        for (int i = tid; i < nv; i += 256)
            *(v4f*)&Hloc[i * 4] = *(const v4f*)&hp[i * 4];
    }
    for (int i = tid; i < NPG * NPG; i += 256) cnt[i] = 0;
    for (int i = tid; i < NPG * heads; i += 256) {
        int nl = i / heads, hd = i % heads;
        esL[nl][hd] = es[(size_t)(g * NPG + nl) * heads + hd];
        edL[nl][hd] = ed[(size_t)(g * NPG + nl) * heads + hd];
    }
    __syncthreads();

    { // count edges (frame-local by construction) + self loops
        int e = g * 256 + tid;
        int s = src[e] - g * NPG;
        int d = dst[e] - g * NPG;
        atomicAdd(&cnt[d * NPG + s], 1);
    }
    if (tid < NPG) atomicAdd(&cnt[tid * NPG + tid], 1);
    __syncthreads();

    // raw leaky-relu scores
    for (int p = tid; p < NPG * NPG; p += 256) {
        int d = p >> 5, s = p & 31;
        for (int hd = 0; hd < heads; ++hd) {
            float sc = 0.f;
            if (cnt[p] > 0) {
                float e0 = esL[s][hd] + edL[d][hd];
                sc = e0 > 0.f ? e0 : 0.2f * e0;
            }
            alpha[p][hd] = sc;
        }
    }
    __syncthreads();

    // per-destination max and count-weighted exp-sum
    if (tid < NPG) {
        for (int hd = 0; hd < heads; ++hd) {
            float mx = -INFINITY;
            for (int s = 0; s < NPG; ++s)
                if (cnt[tid * NPG + s] > 0) {
                    float v = alpha[tid * NPG + s][hd];
                    mx = v > mx ? v : mx;
                }
            if (mx == -INFINITY) mx = 0.f;
            float sum = 0.f;
            for (int s = 0; s < NPG; ++s) {
                int c = cnt[tid * NPG + s];
                if (c > 0) sum += (float)c * __expf(alpha[tid * NPG + s][hd] - mx);
            }
            mL[tid][hd]  = mx;
            wsL[tid][hd] = sum;
        }
    }
    __syncthreads();

    // normalize into dense alpha
    for (int p = tid; p < NPG * NPG; p += 256) {
        int d = p >> 5;
        int c = cnt[p];
        for (int hd = 0; hd < heads; ++hd) {
            float a = 0.f;
            if (c > 0)
                a = (float)c * __expf(alpha[p][hd] - mL[d][hd]) / (wsL[d][hd] + 1e-16f);
            alpha[p][hd] = a;
        }
    }
    __syncthreads();

    // aggregate: out[d,col] = sum_s alpha[d,s,hd(col)] * Hloc[s,col], + epilogue
    const float bnscale = rsqrtf(1.f + 1e-5f);
    for (int idx = tid; idx < NPG * F; idx += 256) {
        int d = idx / F, col = idx % F, hd = col >> 7;
        float acc = 0.f;
        for (int s = 0; s < NPG; ++s)
            acc += alpha[d * NPG + s][hd] * Hloc[s * F + col];
        acc += bias[col];
        acc = acc * (gamma[col] * bnscale) + beta[col];
        if (do_relu) acc = acc > 0.f ? acc : 0.f;
        Xout[(size_t)(g * NPG + d) * F + col] = acc;
    }
}

// ---------------------------------------------------------------------------
// Mean pool over the 32 nodes of each graph
// ---------------------------------------------------------------------------
__global__ __launch_bounds__(128) void k_pool(const float* __restrict__ X,
                                              float* __restrict__ gf)
{
    int t = blockIdx.x, c = threadIdx.x;
    float s = 0.f;
    for (int n = 0; n < NPG; ++n) s += X[(size_t)(t * NPG + n) * 128 + c];
    gf[(size_t)t * 128 + c] = s * (1.f / 32.f);
}

// ---------------------------------------------------------------------------
// Simple transpose: W[R,C] -> WT[C,R]
// ---------------------------------------------------------------------------
__global__ __launch_bounds__(256) void k_transpose(const float* __restrict__ W,
                                                   float* __restrict__ WT,
                                                   int R, int Ccols)
{
    int idx = blockIdx.x * 256 + threadIdx.x;
    if (idx < R * Ccols) {
        int r = idx / Ccols, c = idx % Ccols;
        WT[(size_t)c * R + r] = W[idx];
    }
}

// ---------------------------------------------------------------------------
// One bidirectional LSTM layer. grid.x=2 (0=fwd, 1=bwd), 256 threads.
// pre[t,1024] = x@Wih^T (no bias). h in LDS, c in registers.
// whhT[k*1024+row] = Whh[row][k] -> row-coalesced reads (hot in L2).
// ---------------------------------------------------------------------------
__global__ __launch_bounds__(256) void k_lstm(
    const float* __restrict__ pre_f, const float* __restrict__ pre_b,
    const float* __restrict__ whhT_f, const float* __restrict__ whhT_b,
    const float* __restrict__ b_f, const float* __restrict__ b_b,
    float* __restrict__ hout, int T)
{
    const int dir = blockIdx.x;
    const float* pre  = dir ? pre_b  : pre_f;
    const float* whhT = dir ? whhT_b : whhT_f;
    const float* bb   = dir ? b_b    : b_f;
    const int i = threadIdx.x;

    __shared__ float sh[HID];
    float c = 0.f;
    sh[i] = 0.f;
    __syncthreads();

    for (int step = 0; step < T; ++step) {
        int t = dir ? (T - 1 - step) : step;
        float z[4];
#pragma unroll
        for (int gt = 0; gt < 4; ++gt) {
            int row = gt * HID + i;
            float acc = pre[(size_t)t * 1024 + row] + bb[row];
            for (int k = 0; k < HID; ++k)
                acc += whhT[(size_t)k * 1024 + row] * sh[k];
            z[gt] = acc;
        }
        __syncthreads();
        float ig = 1.f / (1.f + __expf(-z[0]));
        float fg = 1.f / (1.f + __expf(-z[1]));
        float gg = tanhf(z[2]);
        float og = 1.f / (1.f + __expf(-z[3]));
        c = fg * c + ig * gg;
        float h = og * tanhf(c);
        sh[i] = h;
        hout[(size_t)t * 512 + dir * HID + i] = h;
        __syncthreads();
    }
}

// ---------------------------------------------------------------------------
// Attention over time + attended vector. Single block, 1024 threads.
// ---------------------------------------------------------------------------
__global__ __launch_bounds__(1024) void k_attn(
    const float* __restrict__ h1, const float* __restrict__ Wa,
    const float* __restrict__ ba, float* __restrict__ attn_out,
    float* __restrict__ attended)
{
    int t = threadIdx.x;
    __shared__ float sc[TT];
    __shared__ float red[TT];

    float a = ba[0];
    for (int k = 0; k < 512; ++k) a += h1[(size_t)t * 512 + k] * Wa[k];
    red[t] = a;
    __syncthreads();
    for (int s = 512; s > 0; s >>= 1) {
        if (t < s) red[t] = fmaxf(red[t], red[t + s]);
        __syncthreads();
    }
    float mx = red[0];
    __syncthreads();
    float e = __expf(a - mx);
    red[t] = e;
    __syncthreads();
    for (int s = 512; s > 0; s >>= 1) {
        if (t < s) red[t] += red[t + s];
        __syncthreads();
    }
    float at = e / red[0];
    attn_out[t] = at;
    sc[t] = at;
    __syncthreads();
    if (t < 512) {
        float acc = 0.f;
        for (int tt = 0; tt < TT; ++tt) acc += h1[(size_t)tt * 512 + t] * sc[tt];
        attended[t] = acc;
    }
}

// ---------------------------------------------------------------------------
// MLP head: 512 -> 256 -> 128 -> 64 -> 2, BN(eval)+ReLU between. One block.
// ---------------------------------------------------------------------------
__global__ __launch_bounds__(512) void k_mlp(
    const float* __restrict__ attended,
    const float* W1, const float* b1, const float* g1, const float* be1,
    const float* W2, const float* b2, const float* g2, const float* be2,
    const float* W3, const float* b3, const float* g3, const float* be3,
    const float* W4, const float* b4, float* __restrict__ logits)
{
    __shared__ float z0[512], z1[256], z2[128], z3[64];
    int t = threadIdx.x;
    const float bnscale = rsqrtf(1.f + 1e-5f);
    z0[t] = attended[t];
    __syncthreads();
    if (t < 256) {
        float a = b1[t];
        for (int k = 0; k < 512; ++k) a += z0[k] * W1[k * 256 + t];
        a = a * (g1[t] * bnscale) + be1[t];
        z1[t] = a > 0.f ? a : 0.f;
    }
    __syncthreads();
    if (t < 128) {
        float a = b2[t];
        for (int k = 0; k < 256; ++k) a += z1[k] * W2[k * 128 + t];
        a = a * (g2[t] * bnscale) + be2[t];
        z2[t] = a > 0.f ? a : 0.f;
    }
    __syncthreads();
    if (t < 64) {
        float a = b3[t];
        for (int k = 0; k < 128; ++k) a += z2[k] * W3[k * 64 + t];
        a = a * (g3[t] * bnscale) + be3[t];
        z3[t] = a > 0.f ? a : 0.f;
    }
    __syncthreads();
    if (t < 2) {
        float a = b4[t];
        for (int k = 0; k < 64; ++k) a += z3[k] * W4[k * 2 + t];
        logits[t] = a;
    }
}

// ---------------------------------------------------------------------------
// Host-side orchestration
// ---------------------------------------------------------------------------
extern "C" void kernel_launch(void* const* d_in, const int* in_sizes, int n_in,
                              void* d_out, int out_size, void* d_ws, size_t ws_size,
                              hipStream_t stream)
{
    (void)in_sizes; (void)n_in; (void)out_size; (void)ws_size;

    // ---- inputs (setup_inputs dict insertion order) ----
    const float* yolo  = (const float*)d_in[0];
    const float* nodex = (const float*)d_in[1];
    const int*   eidx  = (const int*)  d_in[2];
    const float* Wy  = (const float*)d_in[3];
    const float* by  = (const float*)d_in[4];
    const float* Wg0 = (const float*)d_in[5],  *as0 = (const float*)d_in[6];
    const float* ad0 = (const float*)d_in[7],  *bg0 = (const float*)d_in[8];
    const float* gn0 = (const float*)d_in[9],  *bn0 = (const float*)d_in[10];
    const float* Wg1 = (const float*)d_in[11], *as1 = (const float*)d_in[12];
    const float* ad1 = (const float*)d_in[13], *bg1 = (const float*)d_in[14];
    const float* gn1 = (const float*)d_in[15], *bn1 = (const float*)d_in[16];
    const float* Wg2 = (const float*)d_in[17], *as2 = (const float*)d_in[18];
    const float* ad2 = (const float*)d_in[19], *bg2 = (const float*)d_in[20];
    const float* gn2 = (const float*)d_in[21], *bn2 = (const float*)d_in[22];
    const float* Wgp = (const float*)d_in[23], *bgp = (const float*)d_in[24];
    const float* l0f_Wih = (const float*)d_in[25], *l0f_Whh = (const float*)d_in[26], *l0f_b = (const float*)d_in[27];
    const float* l0b_Wih = (const float*)d_in[28], *l0b_Whh = (const float*)d_in[29], *l0b_b = (const float*)d_in[30];
    const float* l1f_Wih = (const float*)d_in[31], *l1f_Whh = (const float*)d_in[32], *l1f_b = (const float*)d_in[33];
    const float* l1b_Wih = (const float*)d_in[34], *l1b_Whh = (const float*)d_in[35], *l1b_b = (const float*)d_in[36];
    const float* Wa = (const float*)d_in[37], *ba = (const float*)d_in[38];
    const float* W1 = (const float*)d_in[39], *b1 = (const float*)d_in[40];
    const float* g1 = (const float*)d_in[41], *be1 = (const float*)d_in[42];
    const float* W2 = (const float*)d_in[43], *b2 = (const float*)d_in[44];
    const float* g2 = (const float*)d_in[45], *be2 = (const float*)d_in[46];
    const float* W3 = (const float*)d_in[47], *b3 = (const float*)d_in[48];
    const float* g3 = (const float*)d_in[49], *be3 = (const float*)d_in[50];
    const float* W4 = (const float*)d_in[51], *b4 = (const float*)d_in[52];

    // ---- workspace carve (floats) ----
    float* p = (float*)d_ws;
    float* Hbuf  = p;  p += (size_t)NTOT * 512;   // GAT h
    float* Xbuf  = p;  p += (size_t)NTOT * 512;   // GAT activations (ping-pong)
    float* es    = p;  p += (size_t)NTOT * 4;
    float* ed    = p;  p += (size_t)NTOT * 4;
    float* gf    = p;  p += (size_t)TT * 128;
    float* comb  = p;  p += (size_t)TT * 256;
    float* pre_f = p;  p += (size_t)TT * 1024;
    float* pre_b = p;  p += (size_t)TT * 1024;
    float* h0cat = p;  p += (size_t)TT * 512;
    float* h1cat = p;  p += (size_t)TT * 512;
    float* wT0f  = p;  p += (size_t)1024 * 256;
    float* wT0b  = p;  p += (size_t)1024 * 256;
    float* wT1f  = p;  p += (size_t)1024 * 256;
    float* wT1b  = p;  p += (size_t)1024 * 256;
    float* attended = p; p += 512;

    float* out = (float*)d_out;

    // ---- pre-transpose LSTM recurrent weights ----
    k_transpose<<<1024, 256, 0, stream>>>(l0f_Whh, wT0f, 1024, 256);
    k_transpose<<<1024, 256, 0, stream>>>(l0b_Whh, wT0b, 1024, 256);
    k_transpose<<<1024, 256, 0, stream>>>(l1f_Whh, wT1f, 1024, 256);
    k_transpose<<<1024, 256, 0, stream>>>(l1b_Whh, wT1b, 1024, 256);

    dim3 blk(256);
    dim3 gBig((NTOT + 63) / 64, (512 + 63) / 64);
    dim3 gL2 ((NTOT + 63) / 64, (128 + 63) / 64);
    dim3 gPrj((TT + 63) / 64,  (128 + 63) / 64);
    dim3 gPre((TT + 63) / 64,  (1024 + 63) / 64);

    // ---- GAT layer 0 (K=4, zero-padded fallback K-tile) ----
    k_gemm_bf16<<<gBig, blk, 0, stream>>>(nodex, Wg0, nullptr, Hbuf,
                                          NTOT, 512, 4, 4, 512, 512, 0);
    k_attn_coef<<<NTOT, 128, 0, stream>>>(Hbuf, as0, ad0, es, ed, 4);
    k_gat_edge<<<TT, blk, 0, stream>>>(Hbuf, eidx, eidx + EE, es, ed,
                                       bg0, gn0, bn0, Xbuf, 4, 1);

    // ---- GAT layer 1 ----
    k_gemm_bf16<<<gBig, blk, 0, stream>>>(Xbuf, Wg1, nullptr, Hbuf,
                                          NTOT, 512, 512, 512, 512, 512, 0);
    k_attn_coef<<<NTOT, 128, 0, stream>>>(Hbuf, as1, ad1, es, ed, 4);
    k_gat_edge<<<TT, blk, 0, stream>>>(Hbuf, eidx, eidx + EE, es, ed,
                                       bg1, gn1, bn1, Xbuf, 4, 1);

    // ---- GAT layer 2 (1 head, no relu) ----
    k_gemm_bf16<<<gL2, blk, 0, stream>>>(Xbuf, Wg2, nullptr, Hbuf,
                                         NTOT, 128, 512, 512, 128, 128, 0);
    k_attn_coef<<<NTOT, 128, 0, stream>>>(Hbuf, as2, ad2, es, ed, 1);
    k_gat_edge<<<TT, blk, 0, stream>>>(Hbuf, eidx, eidx + EE, es, ed,
                                       bg2, gn2, bn2, Xbuf, 1, 0);

    // ---- graph mean pool + projections into comb[T,256] ----
    k_pool<<<TT, 128, 0, stream>>>(Xbuf, gf);
    k_gemm_bf16<<<gPrj, blk, 0, stream>>>(yolo, Wy, by, comb,
                                          TT, 128, 20, 20, 128, 256, 0);
    k_gemm_bf16<<<gPrj, blk, 0, stream>>>(gf, Wgp, bgp, comb + 128,
                                          TT, 128, 128, 128, 128, 256, 0);

    // ---- LSTM layer 0: pre = comb @ Wih^T (Wih stored [1024,256]) ----
    k_gemm_bf16<<<gPre, blk, 0, stream>>>(comb, l0f_Wih, nullptr, pre_f,
                                          TT, 1024, 256, 256, 256, 1024, 1);
    k_gemm_bf16<<<gPre, blk, 0, stream>>>(comb, l0b_Wih, nullptr, pre_b,
                                          TT, 1024, 256, 256, 256, 1024, 1);
    k_lstm<<<2, 256, 0, stream>>>(pre_f, pre_b, wT0f, wT0b, l0f_b, l0b_b, h0cat, TT);

    // ---- LSTM layer 1 (input 512) ----
    k_gemm_bf16<<<gPre, blk, 0, stream>>>(h0cat, l1f_Wih, nullptr, pre_f,
                                          TT, 1024, 512, 512, 512, 1024, 1);
    k_gemm_bf16<<<gPre, blk, 0, stream>>>(h0cat, l1b_Wih, nullptr, pre_b,
                                          TT, 1024, 512, 512, 512, 1024, 1);
    k_lstm<<<2, 256, 0, stream>>>(pre_f, pre_b, wT1f, wT1b, l1f_b, l1b_b, h1cat, TT);

    // ---- attention pool + MLP head (outputs: logits[0..1], attn[2..1025]) ----
    k_attn<<<1, 1024, 0, stream>>>(h1cat, Wa, ba, out + 2, attended);
    k_mlp<<<1, 512, 0, stream>>>(attended, W1, b1, g1, be1, W2, b2, g2, be2,
                                 W3, b3, g3, be3, W4, b4, out);
}
